// OmniAID_84997402788662
// MI455X (gfx1250) — matrix operand order repaired and checked by reading, not compiled, not run on hardware
//
#include <hip/hip_runtime.h>
#include <hip/hip_bf16.h>
#include <stdint.h>

typedef __bf16 bf16_t;
typedef __attribute__((ext_vector_type(16))) __bf16 v16bf;
typedef __attribute__((ext_vector_type(8)))  __bf16 v8bf;
typedef __attribute__((ext_vector_type(8)))  float  v8f;
typedef __attribute__((ext_vector_type(4)))  unsigned int v4u;
typedef __attribute__((ext_vector_type(4)))  int    v4i;
typedef __attribute__((ext_vector_type(8)))  int    v8i;

#define DEV __device__ __forceinline__

constexpr int   kB    = 32;
constexpr int   kS    = 257;
constexpr int   kD    = 1024;
constexpr int   kH    = 16;
constexpr int   kHD   = 64;
constexpr int   kE    = 8;
constexpr int   kR    = 16;
constexpr int   kGH   = 256;
constexpr int   kN    = kB * kS;
constexpr int   kMT   = (kS + 15) / 16;   // 17 m-tiles of 16 tokens per batch
constexpr int   kSW   = 288;              // padded score width (mult of 32 >= 272)
constexpr float kSCALE = 0.125f;          // 64^-0.5

// ---------------- bf16 helpers (no __bf16 arithmetic; bit-level only) -------
DEV bf16_t bfzero() { union { unsigned short u; bf16_t b; } o; o.u = 0; return o.b; }
DEV bf16_t f2bf(float f) {
  union { float f; unsigned u; } a; a.f = f;
  unsigned r = a.u + 0x7fffu + ((a.u >> 16) & 1u);   // round-to-nearest-even
  union { unsigned short u; bf16_t b; } o; o.u = (unsigned short)(r >> 16);
  return o.b;
}
DEV v8bf v8zero() { v8bf v; for (int i = 0; i < 8; ++i) v[i] = bfzero(); return v; }
DEV v8f  v8fzero() { v8f v; for (int i = 0; i < 8; ++i) v[i] = 0.f; return v; }

// ---------------- Tensor Data Mover: 2-D tile DMA into LDS ------------------
// Builds the D# per CDNA5 ISA 8.3/8.4: group0 = {count=1, lds_addr,
// global_addr[56:0], type=2}; group1 = {data_size=2B, tensor_dim0/1,
// tile_dim0/1, tensor_dim0_stride}. Rows >= rowsValid are OOB -> zero-filled,
// which handles the ragged last m-tile for free. dims/strides in elements.
DEV void tdm_load_2d_bf16(unsigned ldsOff, const void* gptr,
                          unsigned dim0, unsigned rowsValid,
                          unsigned tileRows, unsigned stride) {
  uint64_t ga = (uint64_t)(uintptr_t)gptr;
  v4u g0;
  g0[0] = 1u;                                            // count=1 (valid user D#)
  g0[1] = ldsOff;                                        // lds_addr (bytes)
  g0[2] = (unsigned)ga;                                  // global_addr[31:0]
  g0[3] = (unsigned)((ga >> 32) & 0x01FFFFFFu) | (2u << 30); // addr[56:32] | type=2
  v8i g1;
  g1[0] = (int)(1u << 16);                               // data_size = 2 bytes
  g1[1] = (int)((dim0 & 0xFFFFu) << 16);                 // tensor_dim0[15:0]
  g1[2] = (int)(((dim0 >> 16) & 0xFFFFu) |
                ((rowsValid & 0xFFFFu) << 16));          // dim0 hi | tensor_dim1 lo
  g1[3] = (int)(((rowsValid >> 16) & 0xFFFFu) |
                ((dim0 & 0xFFFFu) << 16));               // dim1 hi | tile_dim0 = dim0
  g1[4] = (int)(tileRows & 0xFFFFu);                     // tile_dim1 (tile_dim2=0)
  g1[5] = (int)stride;                                   // tensor_dim0_stride[31:0]
  g1[6] = 0;                                             // stride hi, dim1_stride lo
  g1[7] = 0;
  v4i z4; for (int i = 0; i < 4; ++i) z4[i] = 0;
#if __clang_major__ >= 23
  v8i z8; for (int i = 0; i < 8; ++i) z8[i] = 0;
  __builtin_amdgcn_tensor_load_to_lds(g0, g1, z4, z4, z8, 0);
#else
  __builtin_amdgcn_tensor_load_to_lds(g0, g1, z4, z4, 0);
#endif
}

// ---------------- WMMA fragment loaders (contiguous k order, same for A and B) --------------
// A fragment: 16(M) x 32(K), row-major source, ld elems per row.
// Lanes 0-15: M=lane, K=0..15; lanes 16-31: M=lane-16, K=16..31.
DEV v16bf load_a(const bf16_t* base, int ld, int lane, bool rowValid) {
  int row = lane & 15;
  int ko  = (lane >> 4) * 16;
  const bf16_t* p = base + (size_t)row * ld + ko;
  v8bf lo = rowValid ? *(const v8bf*)p        : v8zero();
  v8bf hi = rowValid ? *(const v8bf*)(p + 8) : v8zero();
  return __builtin_shufflevector(lo, hi, 0,1,2,3,4,5,6,7,8,9,10,11,12,13,14,15);
}
// B fragment: 32(K) x 16(N); source W stored row-major [N x K] (K contiguous).
DEV v16bf load_b_nk(const bf16_t* W, int ld, int n0, int k0, int lane, bool colValid) {
  int n  = lane & 15;
  int kb = k0 + (lane >> 4) * 16;
  const bf16_t* p = W + (size_t)(n0 + n) * ld + kb;
  v8bf lo = colValid ? *(const v8bf*)p       : v8zero();
  v8bf hi = colValid ? *(const v8bf*)(p + 8) : v8zero();
  return __builtin_shufflevector(lo, hi, 0,1,2,3,4,5,6,7,8,9,10,11,12,13,14,15);
}
// B fragment from K-major source [K x N] (strided gather)
DEV v16bf load_b_kn(const bf16_t* base, int ldk, int lane, int kValid) {
  int n  = lane & 15;
  int kb = (lane >> 4) * 16;
  v16bf r;
  for (int i = 0; i < 16; ++i) {
    int k = kb + i;
    r[i] = (k < kValid) ? base[(size_t)k * ldk + n] : bfzero();
  }
  return r;
}
DEV v8f wmma_bf16(v16bf a, v16bf b, v8f c) {
  return __builtin_amdgcn_wmma_f32_16x16x32_bf16(false, a, false, b, (short)0, c, false, false);
}

// ---------------- elementwise f32 -> bf16 -----------------------------------
__global__ __launch_bounds__(256)
void cvt_f32_bf16(const float* __restrict__ in, bf16_t* __restrict__ out, int n) {
  int i = blockIdx.x * 256 + threadIdx.x;
  if (i < n) out[i] = f2bf(in[i]);
}

// ---------------- gating: pooled mean -> MLP -> top2 -> softmax -------------
__global__ __launch_bounds__(256)
void gating_kernel(const float* __restrict__ X,
                   const float* __restrict__ gw1, const float* __restrict__ gb1,
                   const float* __restrict__ gw2, const float* __restrict__ gb2,
                   float* __restrict__ gates, int* __restrict__ idx) {
  __shared__ float pooled[kD];
  __shared__ float hbuf[kGH];
  __shared__ float lg[kE];
  const int tid = threadIdx.x, b = blockIdx.x;
  for (int j = 0; j < 4; ++j) {
    int d = tid * 4 + j;
    float s = 0.f;
    for (int s0 = 0; s0 < kS; ++s0) s += X[((size_t)(b * kS + s0)) * kD + d];
    pooled[d] = s * (1.f / (float)kS);
  }
  __syncthreads();
  {
    int g = tid;                              // kGH == blockDim
    float acc = gb1[g];
    for (int d = 0; d < kD; ++d) acc += pooled[d] * gw1[(size_t)g * kD + d];
    hbuf[g] = fmaxf(acc, 0.f);
  }
  __syncthreads();
  if (tid < kE) {
    float acc = gb2[tid];
    for (int g = 0; g < kGH; ++g) acc += hbuf[g] * gw2[(size_t)tid * kGH + g];
    lg[tid] = acc;
  }
  __syncthreads();
  if (tid == 0) {
    int i0 = 0;
    for (int e = 1; e < kE; ++e) if (lg[e] > lg[i0]) i0 = e;
    int i1 = (i0 == 0) ? 1 : 0;
    for (int e = 0; e < kE; ++e) if (e != i0 && lg[e] > lg[i1]) i1 = e;
    float g0 = 1.f / (1.f + __expf(lg[i1] - lg[i0]));
    gates[b * 2 + 0] = g0; gates[b * 2 + 1] = 1.f - g0;
    idx[b * 2 + 0] = i0;   idx[b * 2 + 1] = i1;
  }
}

// ---------------- fold gate*S into effective rank-32 factors ----------------
__global__ __launch_bounds__(256)
void prep_expert(const float* __restrict__ U, const float* __restrict__ Sv,
                 const float* __restrict__ V, const float* __restrict__ gates,
                 const int* __restrict__ idx,
                 bf16_t* __restrict__ Vcat, bf16_t* __restrict__ Ueff) {
  const int tid = threadIdx.x, b = blockIdx.x;
  const int e0 = idx[b * 2], e1 = idx[b * 2 + 1];
  const float g0 = gates[b * 2], g1 = gates[b * 2 + 1];
  for (int e = tid; e < 32 * kD; e += 256) {
    int j = e >> 10, d = e & 1023;
    int k = j >> 4, r = j & 15;
    int ex = k ? e1 : e0;
    Vcat[(size_t)b * 32 * kD + e] = f2bf(V[((size_t)ex * kR + r) * kD + d]);
  }
  for (int e = tid; e < kD * 32; e += 256) {
    int o = e >> 5, j = e & 31;
    int k = j >> 4, r = j & 15;
    int ex = k ? e1 : e0;
    float g = k ? g1 : g0;
    Ueff[(size_t)b * kD * 32 + e] =
        f2bf(g * Sv[ex * kR + r] * U[((size_t)ex * kD + o) * kR + r]);
  }
}

// ---------------- XV = X * Vcat^T (N x 32, K = 1024), WMMA ------------------
__global__ __launch_bounds__(64)
void xv_proj(const bf16_t* __restrict__ Xbf, const bf16_t* __restrict__ Vcat,
             bf16_t* __restrict__ XV) {
  __shared__ bf16_t As[16 * kD];
  const int tid = threadIdx.x, lane = tid & 31, w = tid >> 5;
  const int mt = blockIdx.x, b = blockIdx.y;
  const int m0 = mt * 16, n0 = w * 16;
  // TDM stages the 16x1024 A tile; OOB rows (>= kS - m0) zero-fill in HW.
  if (w == 0) {
    tdm_load_2d_bf16((unsigned)(uintptr_t)As,
                     Xbf + ((size_t)(b * kS + m0)) * kD,
                     kD, (unsigned)(kS - m0), 16, kD);
    __builtin_amdgcn_s_wait_tensorcnt(0);
  }
  __syncthreads();
  const bf16_t* Vb = Vcat + (size_t)b * 32 * kD;
  v8f c = v8fzero();
  for (int k0 = 0; k0 < kD; k0 += 32) {
    v16bf a  = load_a(As + k0, kD, lane, true);
    v16bf bb = load_b_nk(Vb, kD, n0, k0, lane, true);
    c = wmma_bf16(a, bb, c);
  }
  const int col = n0 + (lane & 15);
  for (int i = 0; i < 8; ++i) {
    int row = m0 + i + ((lane >> 4) * 8);
    if (row < kS) XV[((size_t)(b * kS + row)) * 32 + col] = f2bf(c[i]);
  }
}

// ---------------- main projection: X*Wm^T + XV*Ueff^T + bias ----------------
__global__ __launch_bounds__(256)
void gemm_moe(const bf16_t* __restrict__ Xbf, const bf16_t* __restrict__ Wbf,
              const bf16_t* __restrict__ XV, const bf16_t* __restrict__ Ueff,
              const float* __restrict__ bias, float scale, int storeBf,
              bf16_t* __restrict__ outBf, float* __restrict__ outF) {
  __shared__ bf16_t As[16 * kD];
  const int tid = threadIdx.x, lane = tid & 31, w = tid >> 5;
  const int b = blockIdx.z, mt = blockIdx.y;
  const int m0 = mt * 16;
  const int n0 = (blockIdx.x * 8 + w) * 16;
  if (w == 0) {
    tdm_load_2d_bf16((unsigned)(uintptr_t)As,
                     Xbf + ((size_t)(b * kS + m0)) * kD,
                     kD, (unsigned)(kS - m0), 16, kD);
    __builtin_amdgcn_s_wait_tensorcnt(0);
  }
  __syncthreads();
  v8f c = v8fzero();
  for (int k0 = 0; k0 < kD; k0 += 32) {
    __builtin_prefetch(Wbf + (size_t)(n0 + (lane & 15)) * kD + k0 + 64, 0, 1);
    v16bf a  = load_a(As + k0, kD, lane, true);
    v16bf bb = load_b_nk(Wbf, kD, n0, k0, lane, true);
    c = wmma_bf16(a, bb, c);
  }
  { // rank-32 expert correction: a single extra WMMA (K = 32)
    const bool rv = (m0 + (lane & 15)) < kS;
    v16bf a  = load_a(XV + (size_t)(b * kS + m0) * 32, 32, lane, rv);
    v16bf bb = load_b_nk(Ueff + (size_t)b * kD * 32, 32, n0, 0, lane, true);
    c = wmma_bf16(a, bb, c);
  }
  const int col = n0 + (lane & 15);
  const float bv = bias[col];
  for (int i = 0; i < 8; ++i) {
    int row = m0 + i + ((lane >> 4) * 8);
    if (row < kS) {
      float val = (c[i] + bv) * scale;
      size_t o = ((size_t)(b * kS + row)) * kD + col;
      if (storeBf) outBf[o] = f2bf(val); else outF[o] = val;
    }
  }
}

// ---------------- attention: per (b, h, m-tile of 16 queries) ---------------
__global__ __launch_bounds__(256)
void attn_kernel(const bf16_t* __restrict__ Q, const bf16_t* __restrict__ K,
                 const bf16_t* __restrict__ V, bf16_t* __restrict__ Ctx) {
  __shared__ float  sc[16][kSW];
  __shared__ bf16_t scb[16 * kSW];
  const int tid = threadIdx.x, lane = tid & 31, w = tid >> 5;
  const int mt = blockIdx.x, h = blockIdx.y, b = blockIdx.z;
  const int m0 = mt * 16;
  const size_t hoff = (size_t)h * kHD;
  const bool rv = (m0 + (lane & 15)) < kS;
  const bf16_t* qb = Q + ((size_t)(b * kS + m0)) * kD + hoff;
  v16bf aq0 = load_a(qb, kD, lane, rv);
  v16bf aq1 = load_a(qb + 32, kD, lane, rv);
  // scores = q . k^T  (SCALE already folded into q)
  for (int nt = w; nt < kMT; nt += 8) {
    int n0 = nt * 16;
    bool cv = (n0 + (lane & 15)) < kS;
    const bf16_t* kb = K + ((size_t)(b * kS + n0)) * kD + hoff;
    v16bf bk0 = load_b_nk(kb, kD, 0, 0, lane, cv);
    v16bf bk1 = load_b_nk(kb + 32, kD, 0, 0, lane, cv);
    v8f c = v8fzero();
    c = wmma_bf16(aq0, bk0, c);
    c = wmma_bf16(aq1, bk1, c);
    for (int i = 0; i < 8; ++i)
      sc[i + ((lane >> 4) * 8)][n0 + (lane & 15)] = c[i];
  }
  __syncthreads();
  // row softmax: 16 lanes per row, cross-lane reductions within half-waves
  {
    int r  = tid >> 4;     // row 0..15
    int cl = tid & 15;     // lane-in-row
    float m = -1e30f;
    for (int col = cl; col < kS; col += 16) m = fmaxf(m, sc[r][col]);
    for (int d = 8; d; d >>= 1) m = fmaxf(m, __shfl_xor(m, d, 32));
    float s = 0.f;
    for (int col = cl; col < kSW; col += 16) {
      float e = (col < kS) ? __expf(sc[r][col] - m) : 0.f;
      sc[r][col] = e; s += e;
    }
    for (int d = 8; d; d >>= 1) s += __shfl_xor(s, d, 32);
    float inv = 1.f / s;
    for (int col = cl; col < kSW; col += 16) scb[r * kSW + col] = f2bf(sc[r][col] * inv);
  }
  __syncthreads();
  // ctx = probs . V  (waves 0..3 cover the 4 head-dim tiles)
  if (w < 4) {
    v8f c = v8fzero();
    for (int kk = 0; kk < kSW; kk += 32) {
      v16bf a = load_a(scb + kk, kSW, lane, true);
      const bf16_t* vb = V + ((size_t)(b * kS + kk)) * kD + hoff + w * 16;
      v16bf bv = load_b_kn(vb, kD, lane, kS - kk);
      c = wmma_bf16(a, bv, c);
    }
    const int col = w * 16 + (lane & 15);
    for (int i = 0; i < 8; ++i) {
      int row = m0 + i + ((lane >> 4) * 8);
      if (row < kS) Ctx[((size_t)(b * kS + row)) * kD + hoff + col] = f2bf(c[i]);
    }
  }
}

// ---------------------------------------------------------------------------
extern "C" void kernel_launch(void* const* d_in, const int* in_sizes, int n_in,
                              void* d_out, int out_size, void* d_ws, size_t ws_size,
                              hipStream_t stream) {
  (void)in_sizes; (void)n_in; (void)out_size; (void)ws_size;
  const float* x   = (const float*)d_in[0];
  const float* gw1 = (const float*)d_in[1];
  const float* gb1 = (const float*)d_in[2];
  const float* gw2 = (const float*)d_in[3];
  const float* gb2 = (const float*)d_in[4];
  const float* Wm[4] = { (const float*)d_in[5],  (const float*)d_in[10],
                         (const float*)d_in[15], (const float*)d_in[20] };
  const float* Um[4] = { (const float*)d_in[6],  (const float*)d_in[11],
                         (const float*)d_in[16], (const float*)d_in[21] };
  const float* Sm[4] = { (const float*)d_in[7],  (const float*)d_in[12],
                         (const float*)d_in[17], (const float*)d_in[22] };
  const float* Vm[4] = { (const float*)d_in[8],  (const float*)d_in[13],
                         (const float*)d_in[18], (const float*)d_in[23] };
  const float* bm[4] = { (const float*)d_in[9],  (const float*)d_in[14],
                         (const float*)d_in[19], (const float*)d_in[24] };

  char* ws = (char*)d_ws; size_t off = 0;
  auto wsalloc = [&](size_t bytes) -> void* {
    void* p = ws + off; off += (bytes + 255) & ~(size_t)255; return p;
  };
  float*  gates = (float*)wsalloc(kB * 2 * sizeof(float));
  int*    idx   = (int*)  wsalloc(kB * 2 * sizeof(int));
  bf16_t* xbf   = (bf16_t*)wsalloc((size_t)kN * kD * 2);
  bf16_t* Wbf[4], *Vcat[4], *Ueff[4], *XV[4];
  for (int p = 0; p < 4; ++p) Wbf[p]  = (bf16_t*)wsalloc((size_t)kD * kD * 2);
  for (int p = 0; p < 4; ++p) Vcat[p] = (bf16_t*)wsalloc((size_t)kB * 32 * kD * 2);
  for (int p = 0; p < 4; ++p) Ueff[p] = (bf16_t*)wsalloc((size_t)kB * kD * 32 * 2);
  for (int p = 0; p < 4; ++p) XV[p]   = (bf16_t*)wsalloc((size_t)kN * 32 * 2);
  bf16_t* qbf   = (bf16_t*)wsalloc((size_t)kN * kD * 2);
  bf16_t* kbf   = (bf16_t*)wsalloc((size_t)kN * kD * 2);
  bf16_t* vbf   = (bf16_t*)wsalloc((size_t)kN * kD * 2);
  bf16_t* ctxbf = (bf16_t*)wsalloc((size_t)kN * kD * 2);

  // 1) convert activations + weights to bf16 once
  {
    int n = kN * kD;
    cvt_f32_bf16<<<dim3((n + 255) / 256), 256, 0, stream>>>(x, xbf, n);
    int nw = kD * kD;
    for (int p = 0; p < 4; ++p)
      cvt_f32_bf16<<<dim3((nw + 255) / 256), 256, 0, stream>>>(Wm[p], Wbf[p], nw);
  }
  // 2) gating
  gating_kernel<<<dim3(kB), 256, 0, stream>>>(x, gw1, gb1, gw2, gb2, gates, idx);
  // 3) fold experts into per-batch rank-32 factors
  for (int p = 0; p < 4; ++p)
    prep_expert<<<dim3(kB), 256, 0, stream>>>(Um[p], Sm[p], Vm[p], gates, idx,
                                              Vcat[p], Ueff[p]);
  // 4) expert-subspace projections for q,k,v (from x)
  for (int p = 0; p < 3; ++p)
    xv_proj<<<dim3(kMT, kB), 64, 0, stream>>>(xbf, Vcat[p], XV[p]);
  // 5) q,k,v projections (bf16 outputs for attention); SCALE folded into q
  gemm_moe<<<dim3(8, kMT, kB), 256, 0, stream>>>(xbf, Wbf[0], XV[0], Ueff[0],
                                                 bm[0], kSCALE, 1, qbf, nullptr);
  gemm_moe<<<dim3(8, kMT, kB), 256, 0, stream>>>(xbf, Wbf[1], XV[1], Ueff[1],
                                                 bm[1], 1.f, 1, kbf, nullptr);
  gemm_moe<<<dim3(8, kMT, kB), 256, 0, stream>>>(xbf, Wbf[2], XV[2], Ueff[2],
                                                 bm[2], 1.f, 1, vbf, nullptr);
  // 6) attention
  attn_kernel<<<dim3(kMT, kH, kB), 256, 0, stream>>>(qbf, kbf, vbf, ctxbf);
  // 7) output projection from ctx -> d_out (f32)
  xv_proj<<<dim3(kMT, kB), 64, 0, stream>>>(ctxbf, Vcat[3], XV[3]);
  gemm_moe<<<dim3(8, kMT, kB), 256, 0, stream>>>(ctxbf, Wbf[3], XV[3], Ueff[3],
                                                 bm[3], 1.f, 0, nullptr,
                                                 (float*)d_out);
}
